// BitFFN_44976897524089
// MI455X (gfx1250) — compile-verified, matrix operand order checked by reading, not run on hardware
//
#include <hip/hip_runtime.h>

// ---------------------------------------------------------------------------
// BitFFN for MI455X (gfx1250, wave32, WMMA + Tensor Data Mover):
//   scale_i = mean(|w_i|);  w_i_q = ternary(w_i, 0.7*scale_i)  (exact in f16)
//   h   = gelu_exact(x @ w1_q^T + b1)     (f16 storage, f32 accumulate)
//   out = h @ w2_q^T + b2                 (f32 output)
// GEMMs: V_WMMA_F32_16X16X32_F16, tiles DMA'd global->LDS by TENSOR_LOAD_TO_LDS
// (double-buffered, TENSORcnt-tracked), TDM inserts the bank-conflict padding.
// ---------------------------------------------------------------------------

typedef _Float16 v16h __attribute__((ext_vector_type(16)));
typedef _Float16 v8h  __attribute__((ext_vector_type(8)));
typedef float    v8f  __attribute__((ext_vector_type(8)));
typedef unsigned int  v4u  __attribute__((ext_vector_type(4)));
typedef int           v4i_ __attribute__((ext_vector_type(4)));
typedef int           v8i_ __attribute__((ext_vector_type(8)));

#define D_MODEL 2048
#define D_FF    8192
#define M_ROWS  (4 * 4096)

#define BM   128
#define BN   128
#define BK   64    // K-chunk per LDS stage (2 wmma k-steps)
#define LDT  72    // LDS row stride in f16 = 64 data + 8 pad (from TDM pad engine)

// ---------------------- absmean reduction ----------------------------------
__global__ __launch_bounds__(256)
void k_abs_sum(const float* __restrict__ w, int n, float* __restrict__ out) {
    __shared__ float red[256];
    float s = 0.0f;
    for (int i = blockIdx.x * blockDim.x + threadIdx.x; i < n;
         i += gridDim.x * blockDim.x)
        s += fabsf(w[i]);
    red[threadIdx.x] = s;
    __syncthreads();
    for (int off = 128; off > 0; off >>= 1) {
        if (threadIdx.x < off) red[threadIdx.x] += red[threadIdx.x + off];
        __syncthreads();
    }
    if (threadIdx.x == 0) atomicAdd(out, red[0]);
}

// ---------------------- ternarize to f16 -----------------------------------
__global__ __launch_bounds__(256)
void k_quant(const float* __restrict__ w, int n, const float* __restrict__ sum,
             _Float16* __restrict__ q) {
    const float thresh = 0.7f * (*sum / (float)n);
    for (int i = blockIdx.x * blockDim.x + threadIdx.x; i < n;
         i += gridDim.x * blockDim.x) {
        float v = w[i];
        float t = (fabsf(v) >= thresh) ? (v > 0.0f ? 1.0f : -1.0f) : 0.0f;
        q[i] = (_Float16)t;
    }
}

// ---------------------- f32 -> f16 cast ------------------------------------
__global__ __launch_bounds__(256)
void k_cast_f16(const float* __restrict__ a, int n, _Float16* __restrict__ o) {
    for (int i = blockIdx.x * blockDim.x + threadIdx.x; i < n;
         i += gridDim.x * blockDim.x)
        o[i] = (_Float16)a[i];
}

// ---------------------- TDM tile load --------------------------------------
__device__ __forceinline__ unsigned rfl(unsigned x) {
    return (unsigned)__builtin_amdgcn_readfirstlane((int)x);
}

// DMA a BM(=128-row) x BK(=64 f16) tile, row stride K elements, into LDS at
// lds_off with 16B of padding appended per row (-> LDS row stride = LDT f16).
__device__ __forceinline__ void tdm_load_tile(const _Float16* __restrict__ g,
                                              unsigned lds_off, int K) {
    unsigned long long ga = (unsigned long long)(uintptr_t)g;
    v4u g0;
    g0[0] = 1u;                                   // count=1 valid descriptor
    g0[1] = rfl(lds_off);                         // lds_addr (bytes)
    g0[2] = rfl((unsigned)ga);                    // global_addr[31:0]
    g0[3] = rfl((unsigned)((ga >> 32) & 0x01FFFFFFull) | (2u << 30)); // [56:32]+type=2
    v8i_ g1;
    // data_size=2B | pad_enable | pad_interval=32dw (code 4) | pad_amount=4dw (code 3)
    g1[0] = (int)((1u << 16) | (1u << 20) | (4u << 22) | (3u << 25));
    g1[1] = (int)((unsigned)(BK & 0xFFFF) << 16); // barrier_addr=0 | tensor_dim0 lo16
    g1[2] = (int)((unsigned)(BM & 0xFFFF) << 16); // tensor_dim0 hi16=0 | tensor_dim1 lo16
    g1[3] = (int)((unsigned)BK << 16);            // tensor_dim1 hi16=0 | tile_dim0=BK
    g1[4] = (int)(unsigned)BM;                    // tile_dim1=BM | tile_dim2=0
    g1[5] = (int)rfl((unsigned)K);                // tensor_dim0_stride[31:0] = K elems
    g1[6] = 0;                                    // stride hi | dim1_stride lo
    g1[7] = 0;
    v4i_ g2 = {0, 0, 0, 0};                       // dims 2/3 unused (2D tile)
    v4i_ g3 = {0, 0, 0, 0};                       // dim 4 unused
    v8i_ g4 = {0, 0, 0, 0, 0, 0, 0, 0};           // extra group (6-arg toolchain form)
    __builtin_amdgcn_tensor_load_to_lds(g0, g1, g2, g3, g4, 0);
}

// Build a v16h A/B fragment per CDNA5 16-bit layout:
//  lanes 0-15 : K = +0..7 (VGPR0-3) and +16..23 (VGPR4-7)
//  lanes16-31 : K = +8..15 and +24..31
__device__ __forceinline__ v16h load_frag(const _Float16* p) {
    v8h lo = *(const v8h*)p;
    v8h hi = *(const v8h*)(p + 16);
    return __builtin_shufflevector(lo, hi, 0, 1, 2, 3, 4, 5, 6, 7,
                                           8, 9, 10, 11, 12, 13, 14, 15);
}

// ---------------------- tiled WMMA GEMM: C = A * Bt^T + bias ---------------
// A: [M,K] f16, Bt: [N,K] f16 (row-major), C: [M,N] OutT.
template <bool GELU, typename OutT>
__global__ __launch_bounds__(256)
void k_gemm_tdm(const _Float16* __restrict__ A, const _Float16* __restrict__ Bt,
                const float* __restrict__ bias, OutT* __restrict__ C,
                int M, int N, int K) {
    __shared__ _Float16 sA[2][BM * LDT];
    __shared__ _Float16 sB[2][BN * LDT];

    const int tid   = threadIdx.x;
    const int wave  = tid >> 5;
    const int lane  = tid & 31;
    const int wm    = wave & 3;       // wave M position: rows wm*32..+32
    const int wn    = wave >> 2;      // wave N position: cols wn*64..+64
    const int l16   = lane & 15;
    const int lhalf = lane >> 4;
    const int kb    = lhalf * 8;      // per-lane K base inside a 32-wide k-step

    const int block_m = blockIdx.y * BM;
    const int block_n = blockIdx.x * BN;

    const _Float16* gA = A  + (size_t)block_m * K;
    const _Float16* gB = Bt + (size_t)block_n * K;

    v8f acc[2][4];
#pragma unroll
    for (int i = 0; i < 2; ++i)
#pragma unroll
        for (int j = 0; j < 4; ++j) acc[i][j] = {};

    const int nk = K / BK;

    // prologue: DMA tile 0 into buffer 0
    if (wave == 0) {
        tdm_load_tile(gA, (unsigned)(uintptr_t)&sA[0][0], K);
        tdm_load_tile(gB, (unsigned)(uintptr_t)&sB[0][0], K);
        __builtin_amdgcn_s_wait_tensorcnt(0);
    }
    __syncthreads();

    int p = 0;
    for (int kt = 0; kt < nk; ++kt) {
        // kick off DMA of the next K-chunk into the other buffer
        if (kt + 1 < nk && wave == 0) {
            const int knext = (kt + 1) * BK;
            tdm_load_tile(gA + knext, (unsigned)(uintptr_t)&sA[p ^ 1][0], K);
            tdm_load_tile(gB + knext, (unsigned)(uintptr_t)&sB[p ^ 1][0], K);
        }

        // compute on current buffer: two 16x16x32 k-steps
#pragma unroll
        for (int ks = 0; ks < BK; ks += 32) {
            v16h afr[2], bfr[4];
#pragma unroll
            for (int i = 0; i < 2; ++i)
                afr[i] = load_frag(&sA[p][(wm * 32 + i * 16 + l16) * LDT + ks + kb]);
#pragma unroll
            for (int j = 0; j < 4; ++j)
                bfr[j] = load_frag(&sB[p][(wn * 64 + j * 16 + l16) * LDT + ks + kb]);
#pragma unroll
            for (int i = 0; i < 2; ++i)
#pragma unroll
                for (int j = 0; j < 4; ++j)
                    acc[i][j] = __builtin_amdgcn_wmma_f32_16x16x32_f16(
                        false, afr[i], false, bfr[j], (short)0, acc[i][j],
                        false, false);
        }

        if (wave == 0) __builtin_amdgcn_s_wait_tensorcnt(0);
        __syncthreads();
        p ^= 1;
    }

    // Epilogue. C/D layout: lanes 0-15 -> rows +0..7 (per VGPR), lanes 16-31 -> +8..15.
#pragma unroll
    for (int i = 0; i < 2; ++i) {
#pragma unroll
        for (int j = 0; j < 4; ++j) {
            const int col  = block_n + wn * 64 + j * 16 + l16;
            const int row0 = block_m + wm * 32 + i * 16 + lhalf * 8;
            const float bv = bias[col];
#pragma unroll
            for (int v = 0; v < 8; ++v) {
                float val = acc[i][j][v] + bv;
                if (GELU)
                    val = 0.5f * val * (1.0f + erff(val * 0.70710678118654752f));
                C[(size_t)(row0 + v) * N + col] = (OutT)val;
            }
        }
    }
}

// ---------------------------------------------------------------------------
extern "C" void kernel_launch(void* const* d_in, const int* in_sizes, int n_in,
                              void* d_out, int out_size, void* d_ws, size_t ws_size,
                              hipStream_t stream) {
    const float* x  = (const float*)d_in[0];
    const float* w1 = (const float*)d_in[1];
    const float* b1 = (const float*)d_in[2];
    const float* w2 = (const float*)d_in[3];
    const float* b2 = (const float*)d_in[4];
    float* out = (float*)d_out;

    const size_t NW = (size_t)D_FF * D_MODEL;     // weight elems (16.7M)
    const size_t NX = (size_t)M_ROWS * D_MODEL;   // x elems (33.5M)

    char* ws = (char*)d_ws;
    float*    sums = (float*)ws;                  // 2 floats
    _Float16* xq   = (_Float16*)(ws + 256);       // [M_ROWS, D_MODEL]
    _Float16* w1q  = xq + NX;                     // [D_FF,   D_MODEL]
    _Float16* w2q  = w1q + NW;                    // [D_MODEL, D_FF]
    _Float16* h    = w2q + NW;                    // [M_ROWS, D_FF]

    (void)hipMemsetAsync(sums, 0, 2 * sizeof(float), stream);
    k_abs_sum<<<1024, 256, 0, stream>>>(w1, (int)NW, sums + 0);
    k_abs_sum<<<1024, 256, 0, stream>>>(w2, (int)NW, sums + 1);
    k_quant<<<2048, 256, 0, stream>>>(w1, (int)NW, sums + 0, w1q);
    k_quant<<<2048, 256, 0, stream>>>(w2, (int)NW, sums + 1, w2q);
    k_cast_f16<<<2048, 256, 0, stream>>>(x, (int)NX, xq);

    dim3 g1(D_FF / BN, M_ROWS / BM);      // 64 x 128 blocks
    k_gemm_tdm<true, _Float16><<<g1, 256, 0, stream>>>(
        xq, w1q, b1, h, M_ROWS, D_FF, D_MODEL);

    dim3 g2(D_MODEL / BN, M_ROWS / BM);   // 16 x 128 blocks
    k_gemm_tdm<false, float><<<g2, 256, 0, stream>>>(
        h, w2q, b2, out, M_ROWS, D_MODEL, D_FF);
}
